// SlidingWindowAttention_67800353734985
// MI455X (gfx1250) — matrix operand order repaired and checked by reading, not compiled, not run on hardware
//
#include <hip/hip_runtime.h>

// Sliding-window attention block for MI455X (gfx1250, wave32, WMMA).
// All GEMM stages use v_wmma_f32_16x16x32_f16 (f16 inputs, f32 accum).

typedef __attribute__((ext_vector_type(16))) _Float16 v16h;
typedef __attribute__((ext_vector_type(8)))  float    v8f;

union Frag {
  v16h h;
  unsigned int u[8];
};

__device__ __forceinline__ v8f wmma_f16(v16h a, v16h b, v8f c) {
  // (neg_a, A, neg_b, B, c_mod, C, reuse_a, reuse_b)
  return __builtin_amdgcn_wmma_f32_16x16x32_f16(false, a, false, b, (short)0, c,
                                                false, false);
}

__device__ __forceinline__ v8f zero8() {
  v8f z = {0.f, 0.f, 0.f, 0.f, 0.f, 0.f, 0.f, 0.f};
  return z;
}

__device__ __forceinline__ int clampi(int v, int lo, int hi) {
  return v < lo ? lo : (v > hi ? hi : v);
}

// ---------------------------------------------------------------------------
// LayerNorm: x[4096,1024] f32 -> xn f16
// ---------------------------------------------------------------------------
__global__ __launch_bounds__(256) void ln_kernel(
    const float* __restrict__ x, const float* __restrict__ gamma,
    const float* __restrict__ beta, _Float16* __restrict__ xn) {
  __shared__ float r1[256];
  __shared__ float r2[256];
  const int row = blockIdx.x;
  const int t = threadIdx.x;
  const float4 v = ((const float4*)(x + (size_t)row * 1024))[t];
  r1[t] = v.x + v.y + v.z + v.w;
  r2[t] = v.x * v.x + v.y * v.y + v.z * v.z + v.w * v.w;
  __syncthreads();
  for (int off = 128; off > 0; off >>= 1) {
    if (t < off) { r1[t] += r1[t + off]; r2[t] += r2[t + off]; }
    __syncthreads();
  }
  const float mu = r1[0] * (1.f / 1024.f);
  const float var = r2[0] * (1.f / 1024.f) - mu * mu;
  const float rs = rsqrtf(var + 1e-5f);
  const int c = t * 4;
  _Float16* o = xn + (size_t)row * 1024 + c;
  o[0] = (_Float16)((v.x - mu) * rs * gamma[c + 0] + beta[c + 0]);
  o[1] = (_Float16)((v.y - mu) * rs * gamma[c + 1] + beta[c + 1]);
  o[2] = (_Float16)((v.z - mu) * rs * gamma[c + 2] + beta[c + 2]);
  o[3] = (_Float16)((v.w - mu) * rs * gamma[c + 3] + beta[c + 3]);
}

// ---------------------------------------------------------------------------
// f32 -> f16 weight conversion
// ---------------------------------------------------------------------------
__global__ __launch_bounds__(256) void cvt_kernel(const float* __restrict__ in,
                                                  _Float16* __restrict__ out,
                                                  int n) {
  int i = blockIdx.x * 256 + threadIdx.x;
  if (i < n) out[i] = (_Float16)in[i];
}

// ---------------------------------------------------------------------------
// Tiled WMMA GEMM: C[M,N] = A[M,K] * W[N,K]^T  (torch Linear convention)
// Block = 256 threads (8 waves), tile 128M x 128N, K-chunk 32.
// MODE 0: QKV epilogue -> scatter q/k f16 [B,H,L,HD] and vT f16 [B,H,HD,L]
// MODE 1: out-proj epilogue -> f32 out + residual
// ---------------------------------------------------------------------------
template <int MODE>
__global__ __launch_bounds__(256) void gemm_wmma(
    const _Float16* __restrict__ A, const _Float16* __restrict__ W, int K,
    _Float16* __restrict__ qf, _Float16* __restrict__ kf,
    _Float16* __restrict__ vtf, const float* __restrict__ residual,
    float* __restrict__ out) {
  __shared__ __align__(16) _Float16 lds_a[128 * 40];  // 32-col tile, stride 40
  __shared__ __align__(16) _Float16 lds_w[128 * 40];

  const int tid = threadIdx.x;
  const int lane = tid & 31;
  const int wid = tid >> 5;
  const int wm = wid >> 2;      // 0..1  -> 64 rows each
  const int wn = wid & 3;       // 0..3  -> 32 cols each
  const int m0 = blockIdx.y * 128;
  const int n0 = blockIdx.x * 128;

  const int r = tid >> 1;           // tile row loaded by this thread
  const int cb = (tid & 1) * 16;    // half-column offset (16 halves)
  const int mlane = lane & 15;
  const int kadd8 = (lane & 16) ? 8 : 0;
  const int kadd16 = (lane & 16) ? 16 : 0;

  v8f acc[4][2];
#pragma unroll
  for (int i = 0; i < 4; ++i)
#pragma unroll
    for (int j = 0; j < 2; ++j) acc[i][j] = zero8();

  for (int kc = 0; kc < K; kc += 32) {
    __syncthreads();
    const uint4* ga = (const uint4*)(A + (size_t)(m0 + r) * K + kc + cb);
    const uint4* gw = (const uint4*)(W + (size_t)(n0 + r) * K + kc + cb);
    uint4 a0 = ga[0], a1 = ga[1];
    uint4 w0 = gw[0], w1 = gw[1];
    *(uint4*)&lds_a[r * 40 + cb] = a0;
    *(uint4*)&lds_a[r * 40 + cb + 8] = a1;
    *(uint4*)&lds_w[r * 40 + cb] = w0;
    *(uint4*)&lds_w[r * 40 + cb + 8] = w1;
    __syncthreads();

    Frag am[4], bn[2];
#pragma unroll
    for (int fm = 0; fm < 4; ++fm) {
      const int row = wm * 64 + fm * 16 + mlane;
#pragma unroll
      for (int v = 0; v < 8; ++v) {
        const int k0 = (v < 4 ? 2 * v : 2 * v + 8) + kadd8;
        am[fm].u[v] = *(const unsigned int*)&lds_a[row * 40 + k0];
      }
    }
#pragma unroll
    for (int fn = 0; fn < 2; ++fn) {
      const int row = wn * 32 + fn * 16 + mlane;
#pragma unroll
      for (int v = 0; v < 8; ++v) {
        const int k0 = (v < 4 ? 2 * v : 2 * v + 8) + kadd8;
        bn[fn].u[v] = *(const unsigned int*)&lds_w[row * 40 + k0];
      }
    }
#pragma unroll
    for (int fm = 0; fm < 4; ++fm)
#pragma unroll
      for (int fn = 0; fn < 2; ++fn)
        acc[fm][fn] = wmma_f16(am[fm].h, bn[fn].h, acc[fm][fn]);
  }

  // Epilogue. C layout: VGPR i -> M = i + 8*(lane>=16), N = lane&15.
#pragma unroll
  for (int fm = 0; fm < 4; ++fm) {
#pragma unroll
    for (int fn = 0; fn < 2; ++fn) {
      const int gn = n0 + wn * 32 + fn * 16 + mlane;
      const int mb = m0 + wm * 64 + fm * 16 + ((lane & 16) ? 8 : 0);
#pragma unroll
      for (int i = 0; i < 8; ++i) {
        const int gm = mb + i;
        const float val = acc[fm][fn][i];
        if (MODE == 0) {
          const int b = gm >> 11, l = gm & 2047;
          const int m3 = gn >> 10, rem = gn & 1023;
          const int hh = rem >> 6, hd = rem & 63;
          const _Float16 hv = (_Float16)val;
          const size_t bh = (size_t)(b * 16 + hh);
          if (m3 == 0)
            qf[(bh * 2048 + l) * 64 + hd] = hv;
          else if (m3 == 1)
            kf[(bh * 2048 + l) * 64 + hd] = hv;
          else
            vtf[(bh * 64 + hd) * 2048 + l] = hv;  // V transposed
        } else {
          const size_t idx = (size_t)gm * 1024 + gn;
          out[idx] = val + residual[idx];
        }
      }
    }
  }
}

// ---------------------------------------------------------------------------
// Banded attention. One wave per (b, h, 16-query tile). blockDim = 128.
// Per-wave dynamic LDS slice: scores f32[16][288], probs f16[16][288],
// inv-rowsum f32[16]  => 27712 bytes (16B aligned).
// ---------------------------------------------------------------------------
__global__ __launch_bounds__(128) void attn_kernel(
    const _Float16* __restrict__ qf, const _Float16* __restrict__ kf,
    const _Float16* __restrict__ vtf, _Float16* __restrict__ ao) {
  extern __shared__ char smem[];
  const int tid = threadIdx.x;
  const int lane = tid & 31;
  const int w = tid >> 5;
  char* base = smem + (size_t)w * 27712;
  float* sc = (float*)base;
  _Float16* pr = (_Float16*)(base + 18432);
  float* rsum = (float*)(base + 18432 + 9216);

  const int task = blockIdx.x * 4 + w;  // 0..4095
  const int qt = task & 127;
  const int hh = (task >> 7) & 15;
  const int b = task >> 11;
  const int q0 = qt * 16;

  const int mlane = lane & 15;
  const int kadd8 = (lane & 16) ? 8 : 0;
  const int kadd16 = (lane & 16) ? 16 : 0;
  const size_t bh = (size_t)(b * 16 + hh);
  const int kb0 = q0 - 128;

  // ---- Q A-fragments over HD=64 (2 K-chunks of 32)
  Frag aq[2];
  {
    const _Float16* qb = qf + (bh * 2048 + (q0 + mlane)) * 64;
#pragma unroll
    for (int c = 0; c < 2; ++c)
#pragma unroll
      for (int v = 0; v < 8; ++v) {
        const int k0 = (v < 4 ? 2 * v : 2 * v + 8) + kadd8 + c * 32;
        aq[c].u[v] = *(const unsigned int*)(qb + k0);
      }
  }

  // ---- scores: 17 key tiles of 16 keys, S = Q*K^T * 1/sqrt(64)
#pragma unroll 1
  for (int t = 0; t < 17; ++t) {
    const int kb = kb0 + t * 16;
    const int keyc = clampi(kb + mlane, 0, 2047);
    const _Float16* kbp = kf + (bh * 2048 + keyc) * 64;
    v8f s = zero8();
#pragma unroll
    for (int c = 0; c < 2; ++c) {
      Frag bk;
#pragma unroll
      for (int v = 0; v < 8; ++v) {
        const int kk0 = (v < 4 ? 2 * v : 2 * v + 8) + kadd8 + c * 32;
        bk.u[v] = *(const unsigned int*)(kbp + kk0);
      }
      s = wmma_f16(aq[c].h, bk.h, s);
    }
#pragma unroll
    for (int i = 0; i < 8; ++i) {
      const int m = i + ((lane & 16) ? 8 : 0);
      const int kj = kb + mlane;
      const int qi = q0 + m;
      const bool ok = (kj >= 0) && (kj < 2048) && (kj >= qi - 128) && (kj <= qi + 128);
      sc[m * 288 + t * 16 + mlane] = ok ? s[i] * 0.125f : -1e30f;
    }
  }
  __syncthreads();

  // ---- softmax (f32): lane pair (l, l+16) owns one row, halves of 272 cols
  {
    const int row = mlane;
    const int c0 = (lane >> 4) * 136;
    const float* srow = sc + row * 288;
    _Float16* prow = pr + row * 288;
    float mloc = -1e30f;
    for (int cc = 0; cc < 136; ++cc) mloc = fmaxf(mloc, srow[c0 + cc]);
    const float mrow = fmaxf(mloc, __shfl_xor(mloc, 16, 32));
    float ssum = 0.f;
    for (int cc = 0; cc < 136; ++cc) {
      const float e = __expf(srow[c0 + cc] - mrow);
      ssum += e;
      prow[c0 + cc] = (_Float16)e;  // unnormalized; divide after PV
    }
    const float tot = ssum + __shfl_xor(ssum, 16, 32);
    if (lane < 16) {
      rsum[row] = 1.0f / tot;
      for (int cc = 272; cc < 288; ++cc) prow[cc] = (_Float16)0.f;  // K pad
    }
  }
  __syncthreads();

  // ---- O = P * V : P is 16x288 (padded), V^T is [HD][L] so B-frag pairs are
  //      contiguous keys. Clamped OOB keys multiply zero probabilities.
#pragma unroll 1
  for (int nc = 0; nc < 4; ++nc) {
    v8f acc = zero8();
    const int hd = nc * 16 + mlane;
    const _Float16* vb = vtf + (bh * 64 + hd) * 2048;
    const _Float16* prow = pr + mlane * 288;
#pragma unroll 1
    for (int c = 0; c < 9; ++c) {
      Frag ap, bv;
#pragma unroll
      for (int v = 0; v < 8; ++v) {
        const int k0 = (v < 4 ? 2 * v : 2 * v + 8) + kadd8 + c * 32;
        ap.u[v] = *(const unsigned int*)(prow + k0);
        const int kk0 = (v < 4 ? 2 * v : 2 * v + 8) + kadd8 + c * 32;
        const int key0 = clampi(kb0 + kk0, 0, 2046);  // even -> b32-aligned pair
        bv.u[v] = *(const unsigned int*)(vb + key0);
      }
      acc = wmma_f16(ap.h, bv.h, acc);
    }
#pragma unroll
    for (int i = 0; i < 8; ++i) {
      const int m = i + ((lane & 16) ? 8 : 0);
      const float o = acc[i] * rsum[m];
      ao[((size_t)(b * 2048 + q0 + m)) * 1024 + hh * 64 + nc * 16 + mlane] =
          (_Float16)o;
    }
  }
}

// ---------------------------------------------------------------------------
// Host launcher
// ---------------------------------------------------------------------------
extern "C" void kernel_launch(void* const* d_in, const int* in_sizes, int n_in,
                              void* d_out, int out_size, void* d_ws,
                              size_t ws_size, hipStream_t stream) {
  (void)in_sizes; (void)n_in; (void)out_size; (void)ws_size;
  const float* x = (const float*)d_in[0];
  const float* wqkv = (const float*)d_in[1];
  const float* wout = (const float*)d_in[2];
  const float* gamma = (const float*)d_in[3];
  const float* beta = (const float*)d_in[4];
  float* out = (float*)d_out;

  // Workspace carve (total ~48 MB, f16 activations)
  char* p = (char*)d_ws;
  _Float16* xn = (_Float16*)p;   p += (size_t)4096 * 1024 * 2;
  _Float16* wqh = (_Float16*)p;  p += (size_t)3072 * 1024 * 2;
  _Float16* woh = (_Float16*)p;  p += (size_t)1024 * 1024 * 2;
  _Float16* qf = (_Float16*)p;   p += (size_t)2 * 16 * 2048 * 64 * 2;
  _Float16* kf = (_Float16*)p;   p += (size_t)2 * 16 * 2048 * 64 * 2;
  _Float16* vtf = (_Float16*)p;  p += (size_t)2 * 16 * 2048 * 64 * 2;
  _Float16* ao = (_Float16*)p;   p += (size_t)4096 * 1024 * 2;

  ln_kernel<<<4096, 256, 0, stream>>>(x, gamma, beta, xn);
  cvt_kernel<<<(3072 * 1024) / 256, 256, 0, stream>>>(wqkv, wqh, 3072 * 1024);
  cvt_kernel<<<(1024 * 1024) / 256, 256, 0, stream>>>(wout, woh, 1024 * 1024);

  // QKV projection: [4096,3072] = xn[4096,1024] x Wqkv[3072,1024]^T
  gemm_wmma<0><<<dim3(24, 32), 256, 0, stream>>>(xn, wqh, 1024, qf, kf, vtf,
                                                 nullptr, nullptr);

  const size_t smem = (size_t)4 * 27712;  // 110,848 B dynamic LDS (<=320KB/WG)
  (void)hipFuncSetAttribute((const void*)attn_kernel,
                            hipFuncAttributeMaxDynamicSharedMemorySize,
                            (int)smem);
  attn_kernel<<<1024, 128, smem, stream>>>(qf, kf, vtf, ao);

  // Out projection + residual: out = ao x Wout^T + x
  gemm_wmma<1><<<dim3(8, 32), 256, 0, stream>>>(ao, woh, 1024, nullptr, nullptr,
                                                nullptr, x, out);
}
